// DeepTempo_46359876993098
// MI455X (gfx1250) — matrix operand order repaired and checked by reading, not compiled, not run on hardware
//
#include <hip/hip_runtime.h>
#include <math.h>

#define NNODES 50000
#define NEDGES 800000
#define HID    32
#define INDIM  64
#define OUTDIM 64
#define NRES   8          // layers 2..9

typedef __attribute__((ext_vector_type(2)))  float    v2f;
typedef __attribute__((ext_vector_type(8)))  float    v8f;
typedef __attribute__((ext_vector_type(16))) _Float16 v16h;

__device__ __forceinline__ float eluf(float x)   { return x > 0.f ? x : expm1f(x); }
__device__ __forceinline__ float sigmf(float x)  { return 1.f / (1.f + expf(-x)); }

// ---------------------------------------------------------------------------
// Generic WMMA GEMM: C[M x N] = A[M x K] * W^T, where W is (N x K) row-major
// (reference weights are stored as W[n][k]; C[m][n] = sum_k A[m][k]*W[n][k]).
// One wave computes one 16x16 tile of C. M,N,K all multiples of 16 here
// (M = 50000 = 3125*16, N in {32,64}, K in {32,64}).
// ---------------------------------------------------------------------------
#if __has_builtin(__builtin_amdgcn_wmma_f32_16x16x4_f32)
// Exact f32 path: V_WMMA_F32_16X16X4_F32.
// A 16x4 layout: lane(0-15) = row M; VGPR0 = K{0 | 2}, VGPR1 = K{1 | 3} (lo|hi half-wave)
// B 4x16 layout: lane(0-15) = col N; VGPR v = K{v | v+2}
// C 16x16     : VGPR v -> M = v + 8*half, N = lane&15
__global__ void wmma_gemm(const float* __restrict__ A, const float* __restrict__ W,
                          float* __restrict__ C, int M, int N, int K)
{
    const int lane   = threadIdx.x & 31;
    const int wave   = threadIdx.x >> 5;
    const int tilesN = N >> 4;
    const int tilesM = M >> 4;
    const int tile   = blockIdx.x * (blockDim.x >> 5) + wave;   // wave-uniform
    if (tile >= tilesM * tilesN) return;
    const int tm   = tile / tilesN;
    const int tn   = tile - tm * tilesN;
    const int half = lane >> 4;     // 0: lanes 0-15, 1: lanes 16-31
    const int l16  = lane & 15;

    const float* arow = A + (size_t)(tm * 16 + l16) * K + 2 * half;
    const float* wrow = W + (size_t)(tn * 16 + l16) * K + 2 * half;

    v8f acc = {};
    for (int k = 0; k < K; k += 4) {
        v2f a, b;
        a[0] = arow[k]; a[1] = arow[k + 1];
        b[0] = wrow[k]; b[1] = wrow[k + 1];
        acc = __builtin_amdgcn_wmma_f32_16x16x4_f32(false, a, false, b,
                                                    (short)0, acc, false, false);
    }
    float* crow = C + (size_t)(tm * 16 + half * 8) * N + tn * 16 + l16;
#pragma unroll
    for (int v = 0; v < 8; ++v) crow[(size_t)v * N] = acc[v];
}
#else
// Fallback: codegen-verified V_WMMA_F32_16X16X32_F16, converting f32->f16 on load.
__global__ void wmma_gemm(const float* __restrict__ A, const float* __restrict__ W,
                          float* __restrict__ C, int M, int N, int K)
{
    const int lane   = threadIdx.x & 31;
    const int wave   = threadIdx.x >> 5;
    const int tilesN = N >> 4;
    const int tilesM = M >> 4;
    const int tile   = blockIdx.x * (blockDim.x >> 5) + wave;
    if (tile >= tilesM * tilesN) return;
    const int tm   = tile / tilesN;
    const int tn   = tile - tm * tilesN;
    const int half = lane >> 4;
    const int l16  = lane & 15;

    const float* arow = A + (size_t)(tm * 16 + l16) * K;
    const float* wrow = W + (size_t)(tn * 16 + l16) * K;

    v8f acc = {};
    for (int k0 = 0; k0 < K; k0 += 32) {
        v16h a, b;
#pragma unroll
        for (int p = 0; p < 8; ++p) {
            // A 16x32: VGPR p<4 -> K = 2p (+8 for hi half); p>=4 -> K = 16+2(p-4) (+8 for hi half)
            int ga = (p < 4) ? ((half ? 8 : 0) + 2 * p) : ((half ? 24 : 16) + 2 * (p - 4));
            a[2 * p]     = (_Float16)arow[k0 + ga];
            a[2 * p + 1] = (_Float16)arow[k0 + ga + 1];
            // B 32x16: same K mapping as A
            int gb = ga;
            b[2 * p]     = (_Float16)wrow[k0 + gb];
            b[2 * p + 1] = (_Float16)wrow[k0 + gb + 1];
        }
        acc = __builtin_amdgcn_wmma_f32_16x16x32_f16(false, a, false, b,
                                                     (short)0, acc, false, false);
    }
    float* crow = C + (size_t)(tm * 16 + half * 8) * N + tn * 16 + l16;
#pragma unroll
    for (int v = 0; v < 8; ++v) crow[(size_t)v * N] = acc[v];
}
#endif

// ---------------------------------------------------------------------------
// Elementwise / graph kernels
// ---------------------------------------------------------------------------
__global__ void zero_f32(float* __restrict__ p, size_t n4)
{
    size_t i = (size_t)blockIdx.x * blockDim.x + threadIdx.x;
    if (i < n4) ((float4*)p)[i] = make_float4(0.f, 0.f, 0.f, 0.f);
}

// One thread per (edge, feature).  Lanes of a wave cover all 32 features of
// one edge: coalesced gather of the source row, coalesced atomics into the
// target row.  unsafeAtomicAdd -> global_atomic_add_f32 (L2-resident).
__global__ void scatter_edges(const int* __restrict__ ei,
                              const float* __restrict__ P,
                              const float* __restrict__ Nn,
                              float* __restrict__ agg)
{
    size_t t = (size_t)blockIdx.x * blockDim.x + threadIdx.x;
    if (t >= (size_t)NEDGES * HID) return;
    int f = (int)(t & (HID - 1));
    int e = (int)(t >> 5);
    int src = ei[3 * e + 0];
    int tgt = ei[3 * e + 1];
    int sgn = ei[3 * e + 2];
    const float* row = (sgn > 0) ? (P + (size_t)src * HID) : (Nn + (size_t)src * HID);
    unsafeAtomicAdd(agg + (size_t)tgt * HID + f, row[f]);
}

// z_new = elu(agg + self);  Z = first ? z_new : z_new + 0.1*Z
__global__ void combine_res(const float* __restrict__ agg, const float* __restrict__ S,
                            float* __restrict__ Z, size_t n, int first)
{
    size_t i = (size_t)blockIdx.x * blockDim.x + threadIdx.x;
    if (i >= n) return;
    float zn = eluf(agg[i] + S[i]);
    Z[i] = first ? zn : (zn + 0.1f * Z[i]);
}

// layernorm over HID=32 columns; one wave per row (wave32)
__global__ void asfr_ln(const float* __restrict__ X, const float* __restrict__ w,
                        const float* __restrict__ b, float* __restrict__ Y)
{
    int r    = (int)(((size_t)blockIdx.x * blockDim.x + threadIdx.x) >> 5);
    int lane = threadIdx.x & 31;
    if (r >= NNODES) return;
    float x = X[(size_t)r * HID + lane];
    float s = x;
    for (int o = 16; o; o >>= 1) s += __shfl_xor(s, o, 32);
    float mu = s * (1.0f / 32.0f);
    float d  = x - mu;
    float q  = d * d;
    for (int o = 16; o; o >>= 1) q += __shfl_xor(q, o, 32);
    float rs = rsqrtf(q * (1.0f / 32.0f) + 1e-5f);
    Y[(size_t)r * HID + lane] = d * rs * w[lane] + b[lane];
}

// ASFR gate + half-swap.  out[r][c] = w1[c]*x[c] + w2[c^16]*x[c^16]
__global__ void asfr_apply(const float* __restrict__ X, const float* __restrict__ G,
                           const float* __restrict__ gb, float* __restrict__ Z)
{
    size_t i = (size_t)blockIdx.x * blockDim.x + threadIdx.x;
    if (i >= (size_t)NNODES * HID) return;
    int c  = (int)(i & (HID - 1));
    size_t rb = i - c;
    int cp = c ^ 16;
    float gc  = sigmf(G[rb + c]  + gb[c]);
    float gp  = sigmf(G[rb + cp] + gb[cp]);
    float w1c = (gc > 0.5f) ? 1.f : gc;
    float w2p = (gp > 0.5f) ? 0.f : gp;
    Z[i] = w1c * X[rb + c] + w2p * X[rb + cp];
}

// z64 = elu(X + bias); write in place AND to d_out (the z output)
__global__ void bias_elu_dual(float* __restrict__ X, const float* __restrict__ b,
                              float* __restrict__ out, size_t n)
{
    size_t i = (size_t)blockIdx.x * blockDim.x + threadIdx.x;
    if (i >= n) return;
    float v = eluf(X[i] + b[(int)(i & (OUTDIM - 1))]);
    X[i] = v;
    out[i] = v;
}

// h = relu(layernorm(X + bias)) over 64 cols; one wave per row (2 cols/lane)
__global__ void bias_ln_relu(const float* __restrict__ X, const float* __restrict__ b,
                             const float* __restrict__ lw, const float* __restrict__ lb,
                             float* __restrict__ Y)
{
    int r    = (int)(((size_t)blockIdx.x * blockDim.x + threadIdx.x) >> 5);
    int lane = threadIdx.x & 31;
    if (r >= NNODES) return;
    size_t base = (size_t)r * OUTDIM;
    float v0 = X[base + lane]      + b[lane];
    float v1 = X[base + 32 + lane] + b[32 + lane];
    float s = v0 + v1;
    for (int o = 16; o; o >>= 1) s += __shfl_xor(s, o, 32);
    float mu = s * (1.0f / 64.0f);
    float d0 = v0 - mu, d1 = v1 - mu;
    float q = d0 * d0 + d1 * d1;
    for (int o = 16; o; o >>= 1) q += __shfl_xor(q, o, 32);
    float rs = rsqrtf(q * (1.0f / 64.0f) + 1e-5f);
    float y0 = d0 * rs * lw[lane]      + lb[lane];
    float y1 = d1 * rs * lw[32 + lane] + lb[32 + lane];
    Y[base + lane]      = fmaxf(y0, 0.f);
    Y[base + 32 + lane] = fmaxf(y1, 0.f);
}

// prob = sigmoid(H . w3 + b3); one wave per row
__global__ void mlp_head(const float* __restrict__ H, const float* __restrict__ w3,
                         const float* __restrict__ b3, float* __restrict__ prob)
{
    int r    = (int)(((size_t)blockIdx.x * blockDim.x + threadIdx.x) >> 5);
    int lane = threadIdx.x & 31;
    if (r >= NNODES) return;
    size_t base = (size_t)r * OUTDIM;
    float s = H[base + lane] * w3[lane] + H[base + 32 + lane] * w3[32 + lane];
    for (int o = 16; o; o >>= 1) s += __shfl_xor(s, o, 32);
    if (lane == 0) prob[r] = sigmf(s + b3[0]);
}

// ---------------------------------------------------------------------------
extern "C" void kernel_launch(void* const* d_in, const int* in_sizes, int n_in,
                              void* d_out, int out_size, void* d_ws, size_t ws_size,
                              hipStream_t stream)
{
    (void)in_sizes; (void)n_in; (void)out_size; (void)ws_size;

    const float* init_emb = (const float*)d_in[0];
    const int*   ei       = (const int*)  d_in[1];
    const float* W1p      = (const float*)d_in[2];
    const float* W1n      = (const float*)d_in[3];
    const float* W1s      = (const float*)d_in[4];
    const float* Wlp      = (const float*)d_in[5];
    const float* Wln      = (const float*)d_in[6];
    const float* Wls      = (const float*)d_in[7];
    const float* a_lnw    = (const float*)d_in[8];
    const float* a_lnb    = (const float*)d_in[9];
    const float* a_gw     = (const float*)d_in[10];
    const float* a_gb     = (const float*)d_in[11];
    const float* proj_w   = (const float*)d_in[12];
    const float* proj_b   = (const float*)d_in[13];
    const float* w1       = (const float*)d_in[14];
    const float* b1       = (const float*)d_in[15];
    const float* ln1w     = (const float*)d_in[16];
    const float* ln1b     = (const float*)d_in[17];
    const float* w2       = (const float*)d_in[18];
    const float* b2       = (const float*)d_in[19];
    const float* ln2w     = (const float*)d_in[20];
    const float* ln2b     = (const float*)d_in[21];
    const float* w3       = (const float*)d_in[22];
    const float* b3       = (const float*)d_in[23];

    float* outz = (float*)d_out;                    // 50000*64
    float* outp = outz + (size_t)NNODES * OUTDIM;   // 50000

    // workspace layout (floats): 50000*320 = 64 MB
    float* Z   = (float*)d_ws;                      // 50000*64 (features; 32 used after L1)
    float* P   = Z   + (size_t)NNODES * 64;         // 50000*32
    float* Nn  = P   + (size_t)NNODES * HID;        // 50000*32
    float* S   = Nn  + (size_t)NNODES * HID;        // 50000*32
    float* AGG = S   + (size_t)NNODES * HID;        // 50000*32
    float* T0  = AGG + (size_t)NNODES * HID;        // 50000*64
    float* T1  = T0  + (size_t)NNODES * 64;         // 50000*64

    const size_t nH  = (size_t)NNODES * HID;        // 1.6M
    const size_t nO  = (size_t)NNODES * OUTDIM;     // 3.2M
    const int ebH  = (int)((nH + 255) / 256);
    const int ebO  = (int)((nO + 255) / 256);
    const int ebZ  = (int)((nH / 4 + 255) / 256);
    const int ebSc = (int)(((size_t)NEDGES * HID + 255) / 256);
    const int ebRw = (int)(((size_t)NNODES * 32 + 255) / 256); // wave-per-row kernels

    auto gemm = [&](const float* A, const float* Wt, float* C, int M, int N, int K) {
        int tiles  = (M / 16) * (N / 16);
        int blocks = (tiles + 7) / 8;               // 8 waves / 256-thread block
        wmma_gemm<<<blocks, 256, 0, stream>>>(A, Wt, C, M, N, K);
    };

    // ---- Layer 1: conv(IN=64 -> HID=32) + ELU + ASFR -----------------------
    gemm(init_emb, W1p, P,  NNODES, HID, INDIM);
    gemm(init_emb, W1n, Nn, NNODES, HID, INDIM);
    gemm(init_emb, W1s, S,  NNODES, HID, INDIM);
    zero_f32<<<ebZ, 256, 0, stream>>>(AGG, nH / 4);
    scatter_edges<<<ebSc, 256, 0, stream>>>(ei, P, Nn, AGG);
    combine_res<<<ebH, 256, 0, stream>>>(AGG, S, T0, nH, 1);
    asfr_ln<<<ebRw, 256, 0, stream>>>(T0, a_lnw, a_lnb, T1);
    gemm(T1, a_gw, P, NNODES, HID, HID);            // gate logits (reuse P)
    asfr_apply<<<ebH, 256, 0, stream>>>(T0, P, a_gb, Z);

    // ---- Layers 2..9: conv(HID -> HID) + ELU + 0.1 residual ----------------
    for (int i = 0; i < NRES; ++i) {
        const float* Wp = Wlp + (size_t)i * HID * HID;
        const float* Wn = Wln + (size_t)i * HID * HID;
        const float* Ws = Wls + (size_t)i * HID * HID;
        gemm(Z, Wp, P,  NNODES, HID, HID);
        gemm(Z, Wn, Nn, NNODES, HID, HID);
        gemm(Z, Ws, S,  NNODES, HID, HID);
        zero_f32<<<ebZ, 256, 0, stream>>>(AGG, nH / 4);
        scatter_edges<<<ebSc, 256, 0, stream>>>(ei, P, Nn, AGG);
        combine_res<<<ebH, 256, 0, stream>>>(AGG, S, Z, nH, (i == 0) ? 1 : 0);
    }

    // ---- Projection + MLP head --------------------------------------------
    gemm(Z, proj_w, T0, NNODES, OUTDIM, HID);
    bias_elu_dual<<<ebO, 256, 0, stream>>>(T0, proj_b, outz, nO);   // z output

    gemm(T0, w1, T1, NNODES, OUTDIM, OUTDIM);
    bias_ln_relu<<<ebRw, 256, 0, stream>>>(T1, b1, ln1w, ln1b, T0);
    gemm(T0, w2, T1, NNODES, OUTDIM, OUTDIM);
    bias_ln_relu<<<ebRw, 256, 0, stream>>>(T1, b2, ln2w, ln2b, T0);
    mlp_head<<<ebRw, 256, 0, stream>>>(T0, w3, b3, outp);           // prob output
}
